// LapLoss_103079215551
// MI455X (gfx1250) — matrix operand order repaired and checked by reading, not compiled, not run on hardware
//
#include <hip/hip_runtime.h>
#include <hip/hip_bf16.h>

typedef __attribute__((ext_vector_type(16))) _Float16 v16h;
typedef __attribute__((ext_vector_type(8)))  float    v8f;

// CDNA5 async global->LDS path (ASYNCcnt-tracked), if this toolchain exposes it.
#if defined(__has_builtin)
#if __has_builtin(__builtin_amdgcn_global_load_async_to_lds_b32) && \
    __has_builtin(__builtin_amdgcn_s_wait_asynccnt)
#define USE_ASYNC_LDS 1
#endif
#endif
#ifndef USE_ASYNC_LDS
#define USE_ASYNC_LDS 0
#endif

typedef __attribute__((address_space(1))) int* as1_int_ptr;
typedef __attribute__((address_space(3))) int* as3_int_ptr;
#define AS_GLOBAL(p) ((as1_int_ptr)(p))
#define AS_LDS(p)    ((as3_int_ptr)(p))

// ---------- index helpers ----------------------------------------------------

// reflect padding in image coordinates: -1 -> 1, H -> H-2 (jnp 'reflect')
__device__ __forceinline__ int refl(int i, int n) {
  i = (i < 0) ? -i : i;
  return (i >= n) ? (2 * n - 2 - i) : i;
}
// reflect of the zero-stuffed image expressed in down-sample coordinates
__device__ __forceinline__ int reflD(int p, int n) {
  p = (p < 0) ? -p : p;
  return (p >= n) ? (2 * n - 1 - p) : p;
}

// CDNA5 16-bit A-matrix (16x32) element->K map, per ISA 7.12.2
__device__ __forceinline__ int a_kidx(int lane, int e) {
  return (lane < 16) ? ((e < 8) ? e : e + 8) : ((e < 8) ? e + 8 : e + 16);
}
// CDNA5 16-bit B-matrix (32x16) element->K map
__device__ __forceinline__ int b_kidx(int lane, int e) {
  return (lane < 16) ? ((e < 8) ? e : e + 8) : ((e < 8) ? e + 8 : e + 16);
}

// Banded coefficient matrices. step=1: plain 5-tap (W[m][k]=g[k-m]).
// step=2: upsample taps on decimated rows (W[y][k]=g4[2k-y]).
__device__ __forceinline__ v16h coefA(int lane, const float* g, int step) {
  v16h a; int M = lane & 15;
#pragma unroll
  for (int e = 0; e < 16; ++e) {
    int d = step * a_kidx(lane, e) - M;
    a[e] = (d >= 0 && d <= 4) ? (_Float16)g[d] : (_Float16)0.0f;
  }
  return a;
}
__device__ __forceinline__ v16h coefB(int lane, const float* g, int step) {
  v16h b; int N = lane & 15;
#pragma unroll
  for (int e = 0; e < 16; ++e) {
    int d = step * b_kidx(lane, e) - N;
    b[e] = (d >= 0 && d <= 4) ? (_Float16)g[d] : (_Float16)0.0f;
  }
  return b;
}

// Gather a 32x16 B operand from an f32 LDS tile (convert during gather).
__device__ __forceinline__ v16h gatherBf(const float* t, int stride, int colOff, int lane) {
  v16h b; int N = (lane & 15) + colOff;
#pragma unroll
  for (int e = 0; e < 16; ++e) b[e] = (_Float16)t[b_kidx(lane, e) * stride + N];
  return b;
}
// Gather a 16x32 A operand from an f16 LDS tile.
__device__ __forceinline__ v16h gatherA(const _Float16* t, int stride, int lane) {
  v16h a; int M = lane & 15;
#pragma unroll
  for (int e = 0; e < 16; ++e) a[e] = t[M * stride + a_kidx(lane, e)];
  return a;
}

// ---------- kernel 1: gauss 5x5 (reflect) + decimate x2 ----------------------
__global__ __launch_bounds__(32) void lap_gauss_down(
    const float* __restrict__ srcA, const float* __restrict__ srcB,
    float* __restrict__ dstA, float* __restrict__ dstB, int H, int W) {
  const float g1[5] = {0.0625f, 0.25f, 0.375f, 0.25f, 0.0625f};
  const int lane = threadIdx.x;
  const int img  = blockIdx.z & 31;
  const int sel  = blockIdx.z >> 5;
  const float* __restrict__ src = (sel ? srcB : srcA) + (size_t)img * H * W;
  const int h = H >> 1, w = W >> 1;
  float* __restrict__ dst = (sel ? dstB : dstA) + (size_t)img * h * w;
  const int Y0 = blockIdx.y * 16, X0 = blockIdx.x * 16;

  __shared__ float    t32[32 * 33]; // 32x32 input halo (f32), padded stride
  __shared__ _Float16 vf[16 * 33];  // 16x32 vertically-filtered tile (f16)

  const int gx = refl(X0 - 2 + lane, W);
#if USE_ASYNC_LDS
#pragma unroll 4
  for (int r = 0; r < 32; ++r) {
    int gy = refl(Y0 - 2 + r, H);
    __builtin_amdgcn_global_load_async_to_lds_b32(
        AS_GLOBAL(src + (size_t)gy * W + gx), AS_LDS(&t32[r * 33 + lane]), 0, 0);
  }
#else
#pragma unroll 4
  for (int r = 0; r < 32; ++r) {
    int gy = refl(Y0 - 2 + r, H);
    t32[r * 33 + lane] = src[(size_t)gy * W + gx];
  }
#endif

  // coefficient operands overlap with the inbound async copy
  v8f  zc = {};
  v16h av = coefA(lane, g1, 1);
  v16h bh = coefB(lane, g1, 1);

#if USE_ASYNC_LDS
  __builtin_amdgcn_s_wait_asynccnt(0);
#endif
  __syncthreads();

  v16h b0 = gatherBf(t32, 33, 0, lane);
  v16h b1 = gatherBf(t32, 33, 16, lane);
  v8f v0 = __builtin_amdgcn_wmma_f32_16x16x32_f16(false, av, false, b0, (short)0, zc, false, false);
  v8f v1 = __builtin_amdgcn_wmma_f32_16x16x32_f16(false, av, false, b1, (short)0, zc, false, false);

  const int N = lane & 15, hi = lane >> 4;
#pragma unroll
  for (int r = 0; r < 8; ++r) {
    vf[(r + 8 * hi) * 33 + N]      = (_Float16)v0[r];
    vf[(r + 8 * hi) * 33 + 16 + N] = (_Float16)v1[r];
  }
  __syncthreads();

  v16h ah = gatherA(vf, 33, lane);
  v8f f = __builtin_amdgcn_wmma_f32_16x16x32_f16(false, ah, false, bh, (short)0, zc, false, false);

  if ((N & 1) == 0) {
#pragma unroll
    for (int r = 0; r < 8; r += 2) {
      int M = r + 8 * hi;  // even
      dst[(size_t)((Y0 + M) >> 1) * w + ((X0 + N) >> 1)] = f[r];
    }
  }
}

// ---------- kernel 2: upsample(down)*4-gauss, diff both tensors, |.| reduce --
__global__ __launch_bounds__(32) void lap_up_diff(
    const float* __restrict__ curA, const float* __restrict__ curB,
    const float* __restrict__ dnA, const float* __restrict__ dnB,
    float* __restrict__ acc, int level, int H, int W) {
  const float g4[5] = {0.125f, 0.5f, 0.75f, 0.5f, 0.125f};
  const int lane = threadIdx.x;
  const int img  = blockIdx.z;
  const int h = H >> 1, w = W >> 1;
  const int Y0 = blockIdx.y * 16, X0 = blockIdx.x * 16;
  const int P0 = Y0 >> 1,  Q0 = X0 >> 1;

  __shared__ float    t32[32 * 17]; // down tile (rows P0-1.., 16 cols), f32
  __shared__ _Float16 vf[16 * 33];  // vertically-upsampled tile (f16)

  const v16h avu = coefA(lane, g4, 2);
  const v16h bhu = coefB(lane, g4, 2);
  const int N = lane & 15, hi = lane >> 4;
  v8f zc = {};
  float dva[8];
  float s = 0.0f;

  for (int sel = 0; sel < 2; ++sel) {
    const float* __restrict__ dn  = (sel ? dnB : dnA) + (size_t)img * h * w;
    const float* __restrict__ cur = (sel ? curB : curA) + (size_t)img * H * W;

    // prefetch the cur tile rows this lane's outputs will touch (read-once data)
    __builtin_prefetch(cur + (size_t)(Y0 + 8 * hi) * W + (X0 + N), 0, 1);

    __syncthreads();
    const int pr = reflD(P0 - 1 + lane, h);
#if USE_ASYNC_LDS
#pragma unroll 4
    for (int c = 0; c < 16; ++c) {
      int qc = reflD(Q0 - 1 + c, w);
      __builtin_amdgcn_global_load_async_to_lds_b32(
          AS_GLOBAL(dn + (size_t)pr * w + qc), AS_LDS(&t32[lane * 17 + c]), 0, 0);
    }
    __builtin_amdgcn_s_wait_asynccnt(0);
#else
#pragma unroll 4
    for (int c = 0; c < 16; ++c) {
      int qc = reflD(Q0 - 1 + c, w);
      t32[lane * 17 + c] = dn[(size_t)pr * w + qc];
    }
#endif
    __syncthreads();

    v16h b = gatherBf(t32, 17, 0, lane);
    v8f vu = __builtin_amdgcn_wmma_f32_16x16x32_f16(false, avu, false, b, (short)0, zc, false, false);

#pragma unroll
    for (int r = 0; r < 8; ++r) {
      vf[(r + 8 * hi) * 33 + N]      = (_Float16)vu[r];
      vf[(r + 8 * hi) * 33 + 16 + N] = (_Float16)0.0f;  // K>=16 taps are zero
    }
    __syncthreads();

    v16h ah = gatherA(vf, 33, lane);
    v8f up = __builtin_amdgcn_wmma_f32_16x16x32_f16(false, ah, false, bhu, (short)0, zc, false, false);

#pragma unroll
    for (int r = 0; r < 8; ++r) {
      int M = r + 8 * hi;
      float c = cur[(size_t)(Y0 + M) * W + (X0 + N)];
      float d = c - up[r];
      if (sel == 0) dva[r] = d;
      else          s += fabsf(dva[r] - d);
    }
  }

  // wave32 reduction + one atomic per wave
#pragma unroll
  for (int off = 16; off > 0; off >>= 1) s += __shfl_down(s, off, 32);
  if (lane == 0) atomicAdd(&acc[level], s);
}

// ---------- tiny helpers -----------------------------------------------------
__global__ void lap_zero_acc(float* acc) {
  if (threadIdx.x < 8) acc[threadIdx.x] = 0.0f;
}
__global__ void lap_finalize(const float* __restrict__ acc, float* __restrict__ out) {
  if (threadIdx.x == 0) {
    float loss = 0.0f;
    int H = 1024;
    for (int l = 0; l < 5; ++l) {
      loss += acc[l] / (32.0f * (float)H * (float)H);
      H >>= 1;
    }
    out[0] = loss;
  }
}

// ---------- host orchestration ----------------------------------------------
extern "C" void kernel_launch(void* const* d_in, const int* in_sizes, int n_in,
                              void* d_out, int out_size, void* d_ws, size_t ws_size,
                              hipStream_t stream) {
  (void)in_sizes; (void)n_in; (void)out_size; (void)ws_size;
  const float* inA = (const float*)d_in[0];
  const float* inB = (const float*)d_in[1];

  // workspace layout: acc[16] | down buffers (both tensors) per level (~90MB, L2-resident)
  float* acc  = (float*)d_ws;
  float* base = acc + 16;
  float* dnA[5]; float* dnB[5];
  size_t off = 0;
  int hs = 512;
  for (int l = 0; l < 5; ++l) {
    dnA[l] = base + off; off += (size_t)32 * hs * hs;
    dnB[l] = base + off; off += (size_t)32 * hs * hs;
    hs >>= 1;
  }

  lap_zero_acc<<<1, 32, 0, stream>>>(acc);

  const float* cA = inA;
  const float* cB = inB;
  int H = 1024;
  for (int l = 0; l < 5; ++l) {
    dim3 gd(H / 16, H / 16, 64);  // z: 32 images x 2 tensors
    lap_gauss_down<<<gd, 32, 0, stream>>>(cA, cB, dnA[l], dnB[l], H, H);
    dim3 gu(H / 16, H / 16, 32);  // z: 32 images (both tensors inside)
    lap_up_diff<<<gu, 32, 0, stream>>>(cA, cB, dnA[l], dnB[l], acc, l, H, H);
    cA = dnA[l]; cB = dnB[l];
    H >>= 1;
  }

  lap_finalize<<<1, 1, 0, stream>>>(acc, (float*)d_out);
}
